// DynamicConvolution2D_79370995630779
// MI455X (gfx1250) — compile-verified, matrix operand order checked by reading, not run on hardware
//
#include <hip/hip_runtime.h>
#include <hip/hip_bf16.h>

// Problem constants (from reference): B=16, T=1024, C=256, H=4, K=31
#define BB   16
#define TT   1024
#define CC   256
#define HH   4
#define KK   31
#define PADK 15
#define MM   (BB * TT)     // 16384 rows
#define NW   (HH * KK)     // 124
#define C2   (2 * CC)      // 512

typedef __attribute__((ext_vector_type(16))) __bf16 v16bf;
typedef __attribute__((ext_vector_type(8)))  __bf16 v8bf;
typedef __attribute__((ext_vector_type(4)))  __bf16 v4bf;
typedef __attribute__((ext_vector_type(8)))  float  v8f;

static __device__ __forceinline__ v8f wmma_bf16(v16bf a, v16bf b, v8f c) {
    // D = A(16x32 bf16) * B(32x16 bf16) + C(16x16 f32)
    return __builtin_amdgcn_wmma_f32_16x16x32_bf16(
        /*neg_a=*/false, a, /*neg_b=*/false, b,
        /*c_mod=*/(short)0, c, /*reuse_a=*/false, /*reuse_b=*/false);
}

// ---- Fragment loaders -------------------------------------------------------
// ISA 16-bit A layout (16x32, wave32): lane row = lane&15, half = lane>>4.
// frag elements 0..7  <- K = k0 + 8*half      + 0..7   (contiguous)
// frag elements 8..15 <- K = k0 + 16 + 8*half + 0..7   (contiguous)
static __device__ __forceinline__ v16bf load_a_bf16(const __bf16* __restrict__ A,
                                                    int lda, int m0, int k0, int lane) {
    int row = m0 + (lane & 15);
    int hl  = lane >> 4;
    const __bf16* p0 = A + (size_t)row * lda + k0 + 8 * hl;   // 16B aligned
    v8bf c0 = *(const v8bf*)p0;          // global_load_b128
    v8bf c1 = *(const v8bf*)(p0 + 16);   // global_load_b128
    v16bf f;
#pragma unroll
    for (int i = 0; i < 8; ++i) { f[i] = c0[i]; f[8 + i] = c1[i]; }
    return f;
}

// B fragments are pre-packed per (kTile,nTile): [32 lanes][16 bf16] contiguous,
// so a fragment load is two contiguous global_load_b128.
static __device__ __forceinline__ v16bf load_b_packed(const __bf16* __restrict__ P,
                                                      int NT, int kt, int nt, int lane) {
    const __bf16* p = P + (((size_t)(kt * NT + nt) * 32 + lane) * 16);
    v8bf c0 = *(const v8bf*)p;
    v8bf c1 = *(const v8bf*)(p + 8);
    v16bf f;
#pragma unroll
    for (int i = 0; i < 8; ++i) { f[i] = c0[i]; f[8 + i] = c1[i]; }
    return f;
}

// ============================================================================
// P0: pack a row-major f32 weight matrix W (Kdim x N) into bf16 WMMA B-fragment
// layout, padding columns to Npad (multiple of 16) with zeros.
// One thread per (tile, lane): 16 strided reads, one 32-byte contiguous write.
// ============================================================================
__global__ void k_pack_b(const float* __restrict__ W, int Kdim, int N, int Npad,
                         __bf16* __restrict__ P) {
    int gid  = blockIdx.x * blockDim.x + threadIdx.x;
    int NT   = Npad >> 4;
    int lane = gid & 31;
    int tile = gid >> 5;
    if (tile >= (Kdim >> 5) * NT) return;
    int kt = tile / NT, nt = tile % NT;
    int col = nt * 16 + (lane & 15);
    int hl  = lane >> 4;
    int k0  = kt * 32;
    v8bf c0, c1;
#pragma unroll
    for (int i = 0; i < 8; ++i) {
        int ka = k0 + 8 * hl + i;
        int kb = ka + 16;
        c0[i] = (col < N) ? (__bf16)W[(size_t)ka * N + col] : (__bf16)0.0f;
        c1[i] = (col < N) ? (__bf16)W[(size_t)kb * N + col] : (__bf16)0.0f;
    }
    __bf16* dst = P + ((size_t)tile * 32 + lane) * 16;
    *(v8bf*)dst       = c0;
    *(v8bf*)(dst + 8) = c1;
}

// ============================================================================
// P1: f32 -> bf16 row-major convert (4 elements/thread).
// ============================================================================
__global__ void k_cvt_bf16(const float* __restrict__ src, __bf16* __restrict__ dst, int n4) {
    int gid = blockIdx.x * blockDim.x + threadIdx.x;
    if (gid >= n4) return;
    float4 v = ((const float4*)src)[gid];
    v4bf o;
    o[0] = (__bf16)v.x; o[1] = (__bf16)v.y; o[2] = (__bf16)v.z; o[3] = (__bf16)v.w;
    ((v4bf*)dst)[gid] = o;
}

// ============================================================================
// K1: x = GLU(query @ w1 + b1); store x as f32 and bf16.
// One wave per 16x64 strip; computes the matching 'a' and 'g' columns
// (8 accumulators, A-fragment reused for 8 WMMAs per 32-deep K step).
// ============================================================================
__global__ void k_glu_gemm(const __bf16* __restrict__ qb,
                           const __bf16* __restrict__ pw1,   // packed, Npad=512
                           const float* __restrict__ b1,
                           float* __restrict__ x_f32,
                           __bf16* __restrict__ x_bf16) {
    int lane = threadIdx.x & 31;
    int job  = blockIdx.x * (blockDim.x >> 5) + (threadIdx.x >> 5);
    int mt   = job >> 2;          // 4 column-groups per 16-row strip
    int g4   = job & 3;
    if (mt >= MM / 16) return;
    int m0 = mt * 16;
    int ntBase = g4 * 4;          // 'a' tiles ntBase..+3 ; 'g' tiles +16
    const int NT = 32;            // 512 packed columns

    v8f acc[8];
#pragma unroll
    for (int i = 0; i < 8; ++i) acc[i] = (v8f){};

    for (int kt = 0; kt < CC / 32; ++kt) {
        v16bf af = load_a_bf16(qb, CC, m0, kt * 32, lane);
#pragma unroll
        for (int i = 0; i < 4; ++i) {
            v16bf ba = load_b_packed(pw1, NT, kt, ntBase + i,      lane);
            acc[i]     = wmma_bf16(af, ba, acc[i]);
            v16bf bg = load_b_packed(pw1, NT, kt, ntBase + i + 16, lane);
            acc[4 + i] = wmma_bf16(af, bg, acc[4 + i]);
        }
    }

    int hl = lane >> 4;
#pragma unroll
    for (int i = 0; i < 4; ++i) {
        int col  = (ntBase + i) * 16 + (lane & 15);
        float ba = b1[col], bg = b1[col + CC];
#pragma unroll
        for (int v = 0; v < 8; ++v) {
            int row  = m0 + v + 8 * hl;
            float a  = acc[i][v] + ba;
            float g  = acc[4 + i][v] + bg;
            float xv = a * (1.0f / (1.0f + __expf(-g)));   // a * sigmoid(g)
            x_f32 [(size_t)row * CC + col] = xv;
            x_bf16[(size_t)row * CC + col] = (__bf16)xv;
        }
    }
}

// ============================================================================
// K2: wcoef = x @ ww + bw (16384x124) and wfk = x @ wf + bf (16384x31).
// 3 jobs per 16-row strip: ww tiles 0-3, ww tiles 4-7, wf tiles 0-1.
// ============================================================================
__global__ void k_coef_gemm(const __bf16* __restrict__ xb,
                            const __bf16* __restrict__ pww,  // packed, Npad=128
                            const __bf16* __restrict__ pwf,  // packed, Npad=32
                            const float* __restrict__ bw,
                            const float* __restrict__ bfv,
                            float* __restrict__ wcoef,
                            float* __restrict__ wfk) {
    int lane = threadIdx.x & 31;
    int job  = blockIdx.x * (blockDim.x >> 5) + (threadIdx.x >> 5);
    int mt = job / 3, sub = job % 3;
    if (mt >= MM / 16) return;
    int m0 = mt * 16;

    const __bf16* P; const float* bias; float* dst;
    int NT, ntBase, ngroup, N;
    if (sub < 2) { P = pww; NT = 8; ntBase = sub * 4; ngroup = 4; N = NW; bias = bw;  dst = wcoef; }
    else         { P = pwf; NT = 2; ntBase = 0;       ngroup = 2; N = KK; bias = bfv; dst = wfk;  }

    v8f acc[4];
#pragma unroll
    for (int i = 0; i < 4; ++i) acc[i] = (v8f){};

    for (int kt = 0; kt < CC / 32; ++kt) {
        v16bf af = load_a_bf16(xb, CC, m0, kt * 32, lane);
#pragma unroll
        for (int i = 0; i < 4; ++i) {
            if (i < ngroup) {
                v16bf bfr = load_b_packed(P, NT, kt, ntBase + i, lane);
                acc[i] = wmma_bf16(af, bfr, acc[i]);
            }
        }
    }

    int hl = lane >> 4;
#pragma unroll
    for (int i = 0; i < 4; ++i) {
        if (i < ngroup) {
            int col = (ntBase + i) * 16 + (lane & 15);
            if (col < N) {
                float bv = bias[col];
#pragma unroll
                for (int v = 0; v < 8; ++v)
                    dst[(size_t)(m0 + v + 8 * hl) * N + col] = acc[i][v] + bv;
            }
        }
    }
}

// ============================================================================
// K3: banded softmax over the 31 relative offsets, in place on wcoef.
// ============================================================================
__global__ void k_softmax(float* __restrict__ wcoef) {
    int gid = blockIdx.x * blockDim.x + threadIdx.x;     // 0 .. B*H*T-1
    if (gid >= BB * HH * TT) return;
    int t  = gid % TT;
    int bh = gid / TT;
    int h  = bh % HH;
    int b  = bh / HH;
    float* p = wcoef + (size_t)(b * TT + t) * NW + h * KK;

    float vals[KK];
    float mx = -1e30f;
#pragma unroll
    for (int j = 0; j < KK; ++j) {
        int s = t - PADK + j;
        bool ok = (s >= 0) && (s < TT);
        float v = ok ? p[j] : -1e30f;
        vals[j] = v;
        mx = fmaxf(mx, v);
    }
    float sum = 0.0f;
#pragma unroll
    for (int j = 0; j < KK; ++j) {
        float e = (vals[j] <= -1e29f) ? 0.0f : __expf(vals[j] - mx);
        vals[j] = e;
        sum += e;
    }
    float inv = 1.0f / sum;
#pragma unroll
    for (int j = 0; j < KK; ++j) p[j] = vals[j] * inv;
}

// ============================================================================
// K4: y (banded attention over time) + xf (31-tap dynamic conv over channels),
// apply mask, store concat [y | xf] as bf16 (16384 x 512) for the final GEMM.
// ============================================================================
__global__ void k_mix(const float* __restrict__ x,
                      const float* __restrict__ attn_all,   // softmaxed wcoef
                      const float* __restrict__ wfk,
                      const int*   __restrict__ mask,
                      __bf16* __restrict__ cat) {
    int m = blockIdx.x;            // 0..MM-1
    int c = threadIdx.x;           // 0..CC-1
    int t = m % TT;
    int b = m / TT;
    int h = c >> 6;                // D = 64 channels per head

    const float* attn = attn_all + (size_t)m * NW + h * KK;
    float y = 0.0f;
#pragma unroll
    for (int j = 0; j < KK; ++j) {
        int s = t - PADK + j;
        if (s >= 0 && s < TT)
            y += attn[j] * x[((size_t)b * TT + s) * CC + c];
    }

    const float* wk = wfk + (size_t)m * KK;
    const float* xr = x   + (size_t)m * CC;
    float xf = 0.0f;
#pragma unroll
    for (int j = 0; j < KK; ++j) {
        int ch = c - PADK + j;
        if (ch >= 0 && ch < CC) xf += wk[j] * xr[ch];
    }

    float mv = (mask[b * TT + t] != 0) ? 1.0f : 0.0f;
    cat[(size_t)m * C2 + c]      = (__bf16)(y  * mv);
    cat[(size_t)m * C2 + CC + c] = (__bf16)(xf * mv);
}

// ============================================================================
// K5: out = cat(16384x512 bf16) @ w2(512x256) + b2 -> d_out (f32).
// One wave per 16x64 strip (4 accumulators), 16 K-steps.
// ============================================================================
__global__ void k_out_gemm(const __bf16* __restrict__ cat,
                           const __bf16* __restrict__ pw2,   // packed, Npad=256
                           const float* __restrict__ b2,
                           float* __restrict__ out) {
    int lane = threadIdx.x & 31;
    int job  = blockIdx.x * (blockDim.x >> 5) + (threadIdx.x >> 5);
    int mt = job >> 2;
    int g4 = job & 3;
    if (mt >= MM / 16) return;
    int m0 = mt * 16;
    int ntBase = g4 * 4;
    const int NT = 16;             // 256 packed columns

    v8f acc[4];
#pragma unroll
    for (int i = 0; i < 4; ++i) acc[i] = (v8f){};

    for (int kt = 0; kt < C2 / 32; ++kt) {
        v16bf af = load_a_bf16(cat, C2, m0, kt * 32, lane);
#pragma unroll
        for (int i = 0; i < 4; ++i) {
            v16bf bfr = load_b_packed(pw2, NT, kt, ntBase + i, lane);
            acc[i] = wmma_bf16(af, bfr, acc[i]);
        }
    }

    int hl = lane >> 4;
#pragma unroll
    for (int i = 0; i < 4; ++i) {
        int col  = (ntBase + i) * 16 + (lane & 15);
        float bv = b2[col];
#pragma unroll
        for (int v = 0; v < 8; ++v)
            out[(size_t)(m0 + v + 8 * hl) * CC + col] = acc[i][v] + bv;
    }
}

// ============================================================================
// Host-side launcher
// ============================================================================
extern "C" void kernel_launch(void* const* d_in, const int* in_sizes, int n_in,
                              void* d_out, int out_size, void* d_ws, size_t ws_size,
                              hipStream_t stream) {
    // Input order: query, key, value, mask, w1, b1, w2, b2, ww, bw, wf, bf
    const float* q    = (const float*)d_in[0];
    const int*   mask = (const int*)  d_in[3];
    const float* w1   = (const float*)d_in[4];
    const float* b1   = (const float*)d_in[5];
    const float* w2   = (const float*)d_in[6];
    const float* b2   = (const float*)d_in[7];
    const float* ww   = (const float*)d_in[8];
    const float* bw   = (const float*)d_in[9];
    const float* wf   = (const float*)d_in[10];
    const float* bfv  = (const float*)d_in[11];
    float* out = (float*)d_out;

    // Workspace layout:
    char* ws = (char*)d_ws;
    size_t off = 0;
    float*  x_f32  = (float*) (ws + off); off += (size_t)MM * CC * sizeof(float);   // 16 MB
    __bf16* x_bf16 = (__bf16*)(ws + off); off += (size_t)MM * CC * sizeof(__bf16);  //  8 MB
    float*  wcoef  = (float*) (ws + off); off += (size_t)MM * NW * sizeof(float);   //  8 MB
    float*  wfk    = (float*) (ws + off); off += (size_t)MM * KK * sizeof(float);   //  2 MB
    __bf16* catb   = (__bf16*)(ws + off); off += (size_t)MM * C2 * sizeof(__bf16);  // 16 MB
    __bf16* qb     = (__bf16*)(ws + off); off += (size_t)MM * CC * sizeof(__bf16);  //  8 MB
    __bf16* pw1    = (__bf16*)(ws + off); off += (size_t)CC * 512 * sizeof(__bf16); // 256 KB
    __bf16* pww    = (__bf16*)(ws + off); off += (size_t)CC * 128 * sizeof(__bf16); //  64 KB
    __bf16* pwf    = (__bf16*)(ws + off); off += (size_t)CC *  32 * sizeof(__bf16); //  16 KB
    __bf16* pw2    = (__bf16*)(ws + off); off += (size_t)C2 * 256 * sizeof(__bf16); // 256 KB
    (void)ws_size; (void)in_sizes; (void)n_in; (void)out_size;

    const int THREADS = 256;                 // 8 waves per block (wave32)
    const int WPB     = THREADS / 32;

    // P0: pack weights into WMMA B-fragment layout (bf16).
    {   // w1: 256x512 -> 8 kt * 32 nt tiles
        int thr = (CC / 32) * (512 / 16) * 32;
        k_pack_b<<<(thr + THREADS - 1) / THREADS, THREADS, 0, stream>>>(w1, CC, C2, 512, pw1);
    }
    {   // ww: 256x124 -> pad to 128
        int thr = (CC / 32) * (128 / 16) * 32;
        k_pack_b<<<(thr + THREADS - 1) / THREADS, THREADS, 0, stream>>>(ww, CC, NW, 128, pww);
    }
    {   // wf: 256x31 -> pad to 32
        int thr = (CC / 32) * (32 / 16) * 32;
        k_pack_b<<<(thr + THREADS - 1) / THREADS, THREADS, 0, stream>>>(wf, CC, KK, 32, pwf);
    }
    {   // w2: 512x256
        int thr = (C2 / 32) * (CC / 16) * 32;
        k_pack_b<<<(thr + THREADS - 1) / THREADS, THREADS, 0, stream>>>(w2, C2, CC, 256, pw2);
    }
    // P1: query f32 -> bf16
    {
        int n4 = MM * CC / 4;
        k_cvt_bf16<<<(n4 + THREADS - 1) / THREADS, THREADS, 0, stream>>>(q, qb, n4);
    }
    // K1: GLU GEMM — 1024 strips * 4 column-groups
    {
        int jobs = (MM / 16) * 4;
        k_glu_gemm<<<(jobs + WPB - 1) / WPB, THREADS, 0, stream>>>(qb, pw1, b1, x_f32, x_bf16);
    }
    // K2: coefficient GEMMs — 1024 strips * 3 jobs
    {
        int jobs = (MM / 16) * 3;
        k_coef_gemm<<<(jobs + WPB - 1) / WPB, THREADS, 0, stream>>>(x_bf16, pww, pwf, bw, bfv, wcoef, wfk);
    }
    // K3: softmax — B*H*T threads
    {
        int n = BB * HH * TT;
        k_softmax<<<(n + THREADS - 1) / THREADS, THREADS, 0, stream>>>(wcoef);
    }
    // K4: attention + dynamic channel conv + mask + concat (bf16)
    {
        k_mix<<<MM, CC, 0, stream>>>(x_f32, wcoef, wfk, mask, catb);
    }
    // K5: output GEMM — 1024 strips * 4 column-groups
    {
        int jobs = (MM / 16) * 4;
        k_out_gemm<<<(jobs + WPB - 1) / WPB, THREADS, 0, stream>>>(catb, pw2, b2, out);
    }
}
